// AttentionBlock_15839839388129
// MI455X (gfx1250) — matrix operand order repaired and checked by reading, not compiled, not run on hardware
//
#include <hip/hip_runtime.h>

typedef _Float16 h16;
typedef _Float16 v16h __attribute__((ext_vector_type(16)));
typedef _Float16 v8h  __attribute__((ext_vector_type(8)));
typedef float    v8f  __attribute__((ext_vector_type(8)));
typedef float    v4f  __attribute__((ext_vector_type(4)));
typedef int      v4i  __attribute__((ext_vector_type(4)));

#define C_CH   512
#define NPIX   4096
#define QSCALE 0.044194173824159216f   /* 512^-0.5 folded into Q */

#if defined(__has_builtin)
#if __has_builtin(__builtin_amdgcn_global_load_async_to_lds_b128) && \
    __has_builtin(__builtin_amdgcn_s_wait_asynccnt)
#define HAVE_ASYNC_LDS 1
typedef __attribute__((address_space(1))) v4i* as1_v4i_p;
typedef __attribute__((address_space(3))) v4i* as3_v4i_p;
#endif
#endif

// ---------------------------------------------------------------- weights->f16
__global__ void wcvt_kernel(const float* __restrict__ wq, const float* __restrict__ wk,
                            const float* __restrict__ wv, const float* __restrict__ wo,
                            h16* __restrict__ dst) {
    int i = blockIdx.x * 256 + threadIdx.x;            // 4 * 262144 total
    if (i >= 4 * C_CH * C_CH) return;
    int mat = i >> 18, off = i & (C_CH * C_CH - 1);
    const float* src = (mat == 0) ? wq : (mat == 1) ? wk : (mat == 2) ? wv : wo;
    dst[i] = (h16)src[off];
}

// ---------------------------------------------------------------- GroupNorm -> hT (pix-major f16)
__global__ void gnorm_kernel(const float* __restrict__ x, const float* __restrict__ gw,
                             const float* __restrict__ gb, h16* __restrict__ hT) {
    int g = blockIdx.x;                 // 0..63 = batch*32 + group
    int bI = g >> 5, grp = g & 31;
    const float* xg = x + (size_t)(bI * C_CH + grp * 16) * NPIX;   // 65536 contiguous f32
    int t = threadIdx.x;
    float s = 0.f, s2 = 0.f;
    for (int i = t; i < 65536 / 4; i += 256) {
        v4f v = ((const v4f*)xg)[i];
        s  += v.x + v.y + v.z + v.w;
        s2 += v.x * v.x + v.y * v.y + v.z * v.z + v.w * v.w;
    }
    __shared__ float r1[256], r2[256];
    r1[t] = s; r2[t] = s2; __syncthreads();
    for (int k = 128; k > 0; k >>= 1) {
        if (t < k) { r1[t] += r1[t + k]; r2[t] += r2[t + k]; }
        __syncthreads();
    }
    float mu   = r1[0] * (1.f / 65536.f);
    float var  = r2[0] * (1.f / 65536.f) - mu * mu;
    float rinv = rsqrtf(var + 1e-5f);
    h16* ho = hT + (size_t)bI * NPIX * C_CH;
    int chbase = grp * 16;
    for (int i = t; i < 65536; i += 256) {
        int ch = chbase + (i >> 12);
        int pix = i & (NPIX - 1);
        float v = (xg[i] - mu) * rinv * gw[ch] + gb[ch];
        ho[(size_t)pix * C_CH + ch] = (h16)v;
    }
}

// ---------------------------------------------------------------- fused QKV projection GEMM
// q,k stored pixel-major (n,c); v stored channel-major (c,n)
__global__ void qkv_kernel(const h16* __restrict__ wAll, const h16* __restrict__ hT,
                           const float* __restrict__ bq, const float* __restrict__ bk,
                           const float* __restrict__ bv,
                           h16* __restrict__ qT, h16* __restrict__ kT, h16* __restrict__ vC) {
    int wave = threadIdx.x >> 5, lane = threadIdx.x & 31;
    int hi = lane >> 4, nn = lane & 15;
    int pz = blockIdx.y, proj = pz >> 1, bI = pz & 1;
    int tile = blockIdx.x * 8 + wave;                 // 8192 16x16 tiles
    int m0 = (tile & 31) << 4;                        // out-channel tile
    int n0 = (tile >> 5) << 4;                        // pixel tile
    const h16* W  = wAll + (size_t)proj * C_CH * C_CH;
    const h16* Bm = hT + (size_t)bI * NPIX * C_CH;
    const h16* arow = W  + (size_t)(m0 + nn) * C_CH + hi * 8;   // A: 16x32 f16 layout
    const h16* bcol = Bm + (size_t)(n0 + nn) * C_CH + hi * 8;   // B: 32x16 f16 layout
    v8f acc = {0.f, 0.f, 0.f, 0.f, 0.f, 0.f, 0.f, 0.f};
    for (int k0 = 0; k0 < C_CH; k0 += 32) {
        v16h a, b;
        ((v8h*)&a)[0] = *(const v8h*)(arow + k0);
        ((v8h*)&a)[1] = *(const v8h*)(arow + k0 + 16);
        ((v8h*)&b)[0] = *(const v8h*)(bcol + k0);
        ((v8h*)&b)[1] = *(const v8h*)(bcol + k0 + 16);
        acc = __builtin_amdgcn_wmma_f32_16x16x32_f16(false, a, false, b, (short)0, acc, false, false);
    }
    if (proj == 2) {
        h16* out = vC + (size_t)bI * C_CH * NPIX;
        for (int r = 0; r < 8; r++) {
            int m = m0 + r + hi * 8;
            out[(size_t)m * NPIX + n0 + nn] = (h16)(acc[r] + bv[m]);
        }
    } else {
        const float* bias = (proj == 0) ? bq : bk;
        float sc = (proj == 0) ? QSCALE : 1.0f;
        h16* out = ((proj == 0) ? qT : kT) + (size_t)bI * NPIX * C_CH;
        for (int r = 0; r < 8; r++) {
            int m = m0 + r + hi * 8;
            out[(size_t)(n0 + nn) * C_CH + m] = (h16)((acc[r] + bias[m]) * sc);
        }
    }
}

// ---------------------------------------------------------------- flash attention
// block = 16 queries; 8 waves; each wave owns a 64-channel slice of the output
__global__ void attn_kernel(const h16* __restrict__ qT, const h16* __restrict__ kT,
                            const h16* __restrict__ vC, h16* __restrict__ oT) {
    __shared__ h16  sQ[16 * 512];     // 16 KB
    __shared__ float sS[16 * 128];    // 8 KB score tile
    __shared__ h16  sP[16 * 128];     // 4 KB prob tile (f16)
    __shared__ float sRed[256];       // reduction scratch
    __shared__ float sM[16], sL[16], sSc[16];
    int t = threadIdx.x;
    int wave = t >> 5, lane = t & 31, hi = lane >> 4, nn = lane & 15;
    int bI = blockIdx.y;
    int q0 = blockIdx.x * 16;
    const h16* Q = qT + (size_t)bI * NPIX * C_CH;
    const h16* K = kT + (size_t)bI * NPIX * C_CH;
    const h16* V = vC + (size_t)bI * C_CH * NPIX;

    {   // stage Q tile into LDS (16 rows x 512 halves = 1024 x 16B)
        const v8h* src = (const v8h*)(Q + (size_t)q0 * C_CH);
        v8h* dst = (v8h*)sQ;
#ifdef HAVE_ASYNC_LDS
        for (int i = t; i < 1024; i += 256)
            __builtin_amdgcn_global_load_async_to_lds_b128(
                (as1_v4i_p)(src + i), (as3_v4i_p)(dst + i), 0, 0);
        __builtin_amdgcn_s_wait_asynccnt(0);
#else
        for (int i = t; i < 1024; i += 256) dst[i] = src[i];
#endif
    }
    if (t < 16) { sM[t] = -1e30f; sL[t] = 0.f; }
    __syncthreads();

    v8f accO[4];
    for (int tt = 0; tt < 4; tt++) accO[tt] = (v8f){0.f,0.f,0.f,0.f,0.f,0.f,0.f,0.f};
    int cbase = wave * 64;

    for (int kt = 0; kt < NPIX / 128; kt++) {
        // ---- S = Q^T K : wave computes key columns [kt*128 + wave*16, +16)
        v8f s = {0.f, 0.f, 0.f, 0.f, 0.f, 0.f, 0.f, 0.f};
        const h16* kcol = K + (size_t)(kt * 128 + wave * 16 + nn) * C_CH + hi * 8;
        const h16* qrow = sQ + nn * 512 + hi * 8;
        if (kt + 1 < NPIX / 128)                  // prefetch next K tile column
            __builtin_prefetch(kcol + 128 * C_CH, 0, 0);
        for (int k0 = 0; k0 < C_CH; k0 += 32) {
            v16h a, b;
            ((v8h*)&a)[0] = *(const v8h*)(qrow + k0);
            ((v8h*)&a)[1] = *(const v8h*)(qrow + k0 + 16);
            ((v8h*)&b)[0] = *(const v8h*)(kcol + k0);
            ((v8h*)&b)[1] = *(const v8h*)(kcol + k0 + 16);
            s = __builtin_amdgcn_wmma_f32_16x16x32_f16(false, a, false, b, (short)0, s, false, false);
        }
        for (int r = 0; r < 8; r++) sS[(r + hi * 8) * 128 + wave * 16 + nn] = s[r];
        __syncthreads();

        // ---- online softmax, parallel over all 256 threads
        {   // stage 1: partial row max (16 threads per row, 8 elements each)
            int row = t >> 4, sub = t & 15;
            float pm = -3e38f;
            for (int j = sub * 8; j < sub * 8 + 8; j++) {
                float v = sS[row * 128 + j];
                pm = v > pm ? v : pm;
            }
            sRed[row * 16 + sub] = pm;
        }
        __syncthreads();
        if (t < 16) {   // stage 2: finalize running max + rescale factor
            float mo = sM[t], mx = mo;
            for (int j = 0; j < 16; j++) { float v = sRed[t * 16 + j]; mx = v > mx ? v : mx; }
            sM[t] = mx;
            sSc[t] = __expf(mo - mx);
        }
        __syncthreads();
        {   // stage 3: exp + f16 P tile + partial row sums
            int row = t >> 4, sub = t & 15;
            float mx = sM[row], sum = 0.f;
            for (int j = sub * 8; j < sub * 8 + 8; j++) {
                float p = __expf(sS[row * 128 + j] - mx);
                sP[row * 128 + j] = (h16)p;
                sum += p;
            }
            sRed[row * 16 + sub] = sum;
        }
        __syncthreads();
        if (t < 16) {   // stage 4: fold into running denominator (overlaps with PV below)
            float sum = 0.f;
            for (int j = 0; j < 16; j++) sum += sRed[t * 16 + j];
            sL[t] = sL[t] * sSc[t] + sum;
        }

        // ---- rescale accumulators, then accO += P * V_slice
        for (int r = 0; r < 8; r++) {
            float sc = sSc[r + hi * 8];
            for (int tt = 0; tt < 4; tt++) accO[tt][r] *= sc;
        }
        for (int kk = 0; kk < 4; kk++) {
            v16h a;
            const h16* prow = sP + nn * 128 + kk * 32 + hi * 8;
            ((v8h*)&a)[0] = *(const v8h*)prow;
            ((v8h*)&a)[1] = *(const v8h*)(prow + 16);
            for (int tt = 0; tt < 4; tt++) {
                v16h b;
                const h16* vcol = V + (size_t)(cbase + tt * 16 + nn) * NPIX
                                    + kt * 128 + kk * 32 + hi * 8;
                ((v8h*)&b)[0] = *(const v8h*)vcol;
                ((v8h*)&b)[1] = *(const v8h*)(vcol + 16);
                accO[tt] = __builtin_amdgcn_wmma_f32_16x16x32_f16(false, a, false, b,
                                                                  (short)0, accO[tt], false, false);
            }
        }
        __syncthreads();
    }
    // ---- normalize and store pixel-major for O-projection
    h16* out = oT + (size_t)bI * NPIX * C_CH;
    for (int r = 0; r < 8; r++) {
        int m = r + hi * 8;
        float inv = 1.f / sL[m];
        for (int tt = 0; tt < 4; tt++)
            out[(size_t)(q0 + m) * C_CH + cbase + tt * 16 + nn] = (h16)(accO[tt][r] * inv);
    }
}

// ---------------------------------------------------------------- O-projection + bias + residual
__global__ void oproj_kernel(const h16* __restrict__ wAll, const h16* __restrict__ oT,
                             const float* __restrict__ bo, const float* __restrict__ x,
                             float* __restrict__ y) {
    int wave = threadIdx.x >> 5, lane = threadIdx.x & 31;
    int hi = lane >> 4, nn = lane & 15;
    int bI = blockIdx.y;
    int tile = blockIdx.x * 8 + wave;
    int m0 = (tile & 31) << 4, n0 = (tile >> 5) << 4;
    const h16* W  = wAll + 3 * (size_t)C_CH * C_CH;
    const h16* Bm = oT + (size_t)bI * NPIX * C_CH;
    const h16* arow = W  + (size_t)(m0 + nn) * C_CH + hi * 8;
    const h16* bcol = Bm + (size_t)(n0 + nn) * C_CH + hi * 8;
    v8f acc = {0.f, 0.f, 0.f, 0.f, 0.f, 0.f, 0.f, 0.f};
    for (int k0 = 0; k0 < C_CH; k0 += 32) {
        v16h a, b;
        ((v8h*)&a)[0] = *(const v8h*)(arow + k0);
        ((v8h*)&a)[1] = *(const v8h*)(arow + k0 + 16);
        ((v8h*)&b)[0] = *(const v8h*)(bcol + k0);
        ((v8h*)&b)[1] = *(const v8h*)(bcol + k0 + 16);
        acc = __builtin_amdgcn_wmma_f32_16x16x32_f16(false, a, false, b, (short)0, acc, false, false);
    }
    for (int r = 0; r < 8; r++) {
        int m = m0 + r + hi * 8;
        size_t idx = ((size_t)bI * C_CH + m) * NPIX + n0 + nn;
        y[idx] = x[idx] + acc[r] + bo[m];
    }
}

// ---------------------------------------------------------------- host launcher
extern "C" void kernel_launch(void* const* d_in, const int* in_sizes, int n_in,
                              void* d_out, int out_size, void* d_ws, size_t ws_size,
                              hipStream_t stream) {
    (void)in_sizes; (void)n_in; (void)out_size; (void)ws_size;
    const float* x   = (const float*)d_in[0];
    const float* gnw = (const float*)d_in[1];
    const float* gnb = (const float*)d_in[2];
    const float* wq  = (const float*)d_in[3];
    const float* bq  = (const float*)d_in[4];
    const float* wk  = (const float*)d_in[5];
    const float* bk  = (const float*)d_in[6];
    const float* wv  = (const float*)d_in[7];
    const float* bv  = (const float*)d_in[8];
    const float* wo  = (const float*)d_in[9];
    const float* bo  = (const float*)d_in[10];

    char* ws = (char*)d_ws;
    const size_t MB = 1024 * 1024;
    h16* wAll = (h16*)(ws);               // 2 MB : wq|wk|wv|wo f16
    h16* hT   = (h16*)(ws + 2  * MB);     // 8 MB : groupnorm out, (b,n,c)
    h16* qT   = (h16*)(ws + 10 * MB);     // 8 MB : (b,n,c), pre-scaled
    h16* kT   = (h16*)(ws + 18 * MB);     // 8 MB : (b,n,c)
    h16* vC   = (h16*)(ws + 26 * MB);     // 8 MB : (b,c,n)
    h16* oT   = (h16*)(ws + 34 * MB);     // 8 MB : attention out (b,n,c)

    wcvt_kernel <<<4096, 256, 0, stream>>>(wq, wk, wv, wo, wAll);
    gnorm_kernel<<<64,   256, 0, stream>>>(x, gnw, gnb, hT);
    qkv_kernel  <<<dim3(1024, 6), 256, 0, stream>>>(wAll, hT, bq, bk, bv, qT, kT, vC);
    attn_kernel <<<dim3(256,  2), 256, 0, stream>>>(qT, kT, vC, oT);
    oproj_kernel<<<dim3(1024, 2), 256, 0, stream>>>(wAll, oT, bo, x, (float*)d_out);
}
